// AttentionDecoder_35296041239158
// MI455X (gfx1250) — matrix operand order repaired and checked by reading, not compiled, not run on hardware
//
#include <hip/hip_runtime.h>
#include <hip/hip_bf16.h>

#define B_  32
#define S_  4096
#define H_  256
#define D_  512
#define KC_ 16   // 512 / 32 k-chunks
#define NT_ 32   // 512 / 16 n-tiles
#define WPAD 520 // 512 + 8 bf16 pad -> row stride 1040B, conflict-free ds_load_b128

typedef __attribute__((ext_vector_type(16))) __bf16 bf16x16;
typedef __attribute__((ext_vector_type(8)))  __bf16 bf16x8;
typedef __attribute__((ext_vector_type(8)))  float  f32x8;

__device__ __forceinline__ void wait_async0() {
#if defined(__has_builtin)
#if __has_builtin(__builtin_amdgcn_s_wait_asynccnt)
    __builtin_amdgcn_s_wait_asynccnt(0);
#else
    asm volatile("s_wait_asynccnt 0x0" ::: "memory");
#endif
#else
    asm volatile("s_wait_asynccnt 0x0" ::: "memory");
#endif
}

// issue one 16-byte async global->LDS transfer (ASYNCcnt tracked)
__device__ __forceinline__ void async_copy_b128(const void* gsrc, void* ldst) {
    const unsigned long long ga = (unsigned long long)(uintptr_t)gsrc;
    const unsigned int       la = (unsigned int)(uintptr_t)ldst; // flat LDS addr[31:0] == LDS offset
    asm volatile("global_load_async_to_lds_b128 %0, %1, off"
                 :: "v"(la), "v"(ga) : "memory");
}

// ---------------- kernel 0: Wh f32 -> bf16 (row-major, contiguous) ----------------
__global__ __launch_bounds__(256) void whcvt_kernel(
    const float* __restrict__ Wh, __bf16* __restrict__ whbf)
{
    const int i = (blockIdx.x * 256 + threadIdx.x) * 4;
    const float4 w4 = *(const float4*)(Wh + i);
    __bf16* d = whbf + i;
    d[0]=(__bf16)w4.x; d[1]=(__bf16)w4.y; d[2]=(__bf16)w4.z; d[3]=(__bf16)w4.w;
}

// ---------------- kernel 1: dec_feat[b,e] = s_t[b,:] . Ws[e,:] + b[e] ----------------
__global__ __launch_bounds__(256) void decfeat_kernel(
    const float* __restrict__ hdec, const float* __restrict__ cdec,
    const float* __restrict__ Ws,   const float* __restrict__ Wsb,
    float* __restrict__ dfeat)
{
    __shared__ float st[D_];
    const int b = blockIdx.x, tid = threadIdx.x;
    st[tid]       = hdec[(size_t)b * H_ + tid];   // concat(h, c) along feature dim
    st[tid + H_]  = cdec[(size_t)b * H_ + tid];
    __syncthreads();
    for (int e = tid; e < D_; e += 256) {
        float acc = Wsb[e];
        const float* wr = Ws + (size_t)e * D_;
        for (int d = 0; d < D_; ++d) acc = fmaf(st[d], wr[d], acc);
        dfeat[(size_t)b * D_ + e] = acc;
    }
}

// ---------------- kernel 2: fused GEMM + tanh + dot(v) -> score ----------------
// block = 256 threads = 8 waves; wave owns 16 rows; block owns 128 rows (never crosses batch).
// W tiles stream global->LDS via async copies, double-buffered against the WMMA loop.
__global__ __launch_bounds__(256) void score_kernel(
    const float*  __restrict__ enc,   // (B,S,D)
    const __bf16* __restrict__ whbf,  // (D,D) bf16: row e, col d
    const float*  __restrict__ dfeat, // (B,D)
    const float*  __restrict__ cov,   // (B,S)
    const float*  __restrict__ wc,    // (D)
    const float*  __restrict__ vw,    // (D)
    float* __restrict__ score)        // (B,S) flat
{
    __shared__ __align__(16) __bf16 wbuf[2][16][WPAD]; // double-buffered 16xD_ bf16 W tiles

    const int tid  = threadIdx.x;
    const int lane = tid & 31;
    const int wave = tid >> 5;
    const int l16  = lane & 15;
    const int halfSel = lane >> 4;                 // 0: lanes 0-15, 1: lanes 16-31

    const int rowTileBase = blockIdx.x * 128;
    const int waveRowBase = rowTileBase + wave * 16;
    const int b = rowTileBase / S_;

    // issue async copy of one 16xD_ W tile into wbuf[pb]: 1024 b128 chunks / 256 threads
    auto issue_wtile = [&](int nt, int pb) {
        #pragma unroll
        for (int it = 0; it < 4; ++it) {
            const int idx  = tid + it * 256;
            const int nrow = idx >> 6;         // 64 chunks per row
            const int c8   = (idx & 63) << 3;  // bf16 column
            async_copy_b128(whbf + (size_t)(nt * 16 + nrow) * D_ + c8,
                            &wbuf[pb][nrow][c8]);
        }
    };

    // kick tile 0 while we load A fragments from global
    issue_wtile(0, 0);

    // ---- load this wave's 16xK A-tile once, f32 -> bf16, into WMMA A-fragment layout ----
    // lanes 0-15 (half 0): K = k0+[0..7] and k0+[16..23] ; lanes 16-31: +8 / +24
    const int myRow = waveRowBase + l16;
    const int ka0 = halfSel * 8;
    const int ka1 = 16 + halfSel * 8;
    const float* arow = enc + (size_t)myRow * D_;

    bf16x16 afrag[KC_];
    #pragma unroll
    for (int kc = 0; kc < KC_; ++kc) {
        const float4 q0 = *(const float4*)(arow + kc * 32 + ka0);
        const float4 q1 = *(const float4*)(arow + kc * 32 + ka0 + 4);
        const float4 q2 = *(const float4*)(arow + kc * 32 + ka1);
        const float4 q3 = *(const float4*)(arow + kc * 32 + ka1 + 4);
        bf16x16 f;
        f[0]=(__bf16)q0.x; f[1]=(__bf16)q0.y; f[2]=(__bf16)q0.z; f[3]=(__bf16)q0.w;
        f[4]=(__bf16)q1.x; f[5]=(__bf16)q1.y; f[6]=(__bf16)q1.z; f[7]=(__bf16)q1.w;
        f[8]=(__bf16)q2.x; f[9]=(__bf16)q2.y; f[10]=(__bf16)q2.z; f[11]=(__bf16)q2.w;
        f[12]=(__bf16)q3.x; f[13]=(__bf16)q3.y; f[14]=(__bf16)q3.z; f[15]=(__bf16)q3.w;
        afrag[kc] = f;
    }

    // per-accumulator-row coverage (C layout: VGPR r -> M = r + 8*halfSel)
    float covr[8];
    #pragma unroll
    for (int r = 0; r < 8; ++r)
        covr[r] = cov[waveRowBase + halfSel * 8 + r];

    float sacc[8] = {0.f,0.f,0.f,0.f,0.f,0.f,0.f,0.f};

    // tile 0 resident before first use
    wait_async0();
    __syncthreads();

    int p = 0;
    for (int nt = 0; nt < NT_; ++nt) {
        // prefetch next W tile into the other buffer (its readers finished last iteration)
        if (nt + 1 < NT_) issue_wtile(nt + 1, p ^ 1);

        f32x8 acc = {};
        #pragma unroll
        for (int kc = 0; kc < KC_; ++kc) {
            // B-fragment (16-bit B 32x16): lanes 0-15 K=k0+[0..7],[16..23]; lanes 16-31 K=k0+[8..15],[24..31]
            const __bf16* wp = &wbuf[p][l16][kc * 32 + halfSel * 8];
            const bf16x8 b0 = *(const bf16x8*)(wp);
            const bf16x8 b1 = *(const bf16x8*)(wp + 16);
            bf16x16 bf;
            #pragma unroll
            for (int i = 0; i < 8; ++i) { bf[i] = b0[i]; bf[8 + i] = b1[i]; }
            acc = __builtin_amdgcn_wmma_f32_16x16x32_bf16(
                false, afrag[kc], false, bf, (short)0, acc, false, false);
        }

        // fused epilogue: + dec_feat + cov*wc -> tanh -> * v, accumulate per row
        const int ncol = nt * 16 + l16;
        const float df  = dfeat[(size_t)b * D_ + ncol];
        const float wcn = wc[ncol];
        const float vn  = vw[ncol];
        #pragma unroll
        for (int r = 0; r < 8; ++r) {
            const float x = acc[r] + df + covr[r] * wcn;
            sacc[r] += vn * tanhf(x);
        }

        wait_async0();      // next tile's async writes to LDS complete
        __syncthreads();    // all waves done (reads of wbuf[p] + writes of wbuf[p^1])
        p ^= 1;
    }

    // reduce over the 16 columns held across lanes of each half
    #pragma unroll
    for (int r = 0; r < 8; ++r) {
        float v = sacc[r];
        v += __shfl_xor(v, 8, 32);
        v += __shfl_xor(v, 4, 32);
        v += __shfl_xor(v, 2, 32);
        v += __shfl_xor(v, 1, 32);
        if (l16 == 0) score[waveRowBase + halfSel * 8 + r] = v;
    }
}

// ---------------- kernel 3: masked softmax + coverage update ----------------
__global__ __launch_bounds__(256) void softmax_kernel(
    const float* __restrict__ score, const float* __restrict__ mask,
    const float* __restrict__ cov, float* __restrict__ attn,
    float* __restrict__ covn)
{
    __shared__ float redA[8];
    __shared__ float redB[8];
    const int b = blockIdx.x, tid = threadIdx.x;
    const size_t base = (size_t)b * S_;

    float v[16];
    float lmax = -3.402823466e38f;
    #pragma unroll
    for (int i = 0; i < 16; ++i) {
        v[i] = score[base + tid + i * 256];
        lmax = fmaxf(lmax, v[i]);
    }
    #pragma unroll
    for (int off = 16; off > 0; off >>= 1) lmax = fmaxf(lmax, __shfl_xor(lmax, off, 32));
    if ((tid & 31) == 0) redA[tid >> 5] = lmax;
    __syncthreads();
    float bmax = redA[0];
    #pragma unroll
    for (int w = 1; w < 8; ++w) bmax = fmaxf(bmax, redA[w]);

    float lsum = 0.f;
    #pragma unroll
    for (int i = 0; i < 16; ++i) {
        const float e = __expf(v[i] - bmax) * mask[base + tid + i * 256];
        v[i] = e;
        lsum += e;
    }
    #pragma unroll
    for (int off = 16; off > 0; off >>= 1) lsum += __shfl_xor(lsum, off, 32);
    if ((tid & 31) == 0) redB[tid >> 5] = lsum;
    __syncthreads();
    float tot = 0.f;
    #pragma unroll
    for (int w = 0; w < 8; ++w) tot += redB[w];
    const float inv = 1.0f / tot;

    #pragma unroll
    for (int i = 0; i < 16; ++i) {
        const float a = v[i] * inv;
        attn[base + tid + i * 256] = a;
        covn[base + tid + i * 256] = cov[base + tid + i * 256] + a;
    }
}

// ---------------- kernel 4: context partials (split-S GEMV) ----------------
__global__ __launch_bounds__(256) void ctx_partial_kernel(
    const float* __restrict__ attn, const float* __restrict__ enc,
    float* __restrict__ part)
{
    __shared__ float al[512];
    const int b = blockIdx.x, c = blockIdx.y, tid = threadIdx.x;
    const int s0 = c * 512;
    al[tid]       = attn[(size_t)b * S_ + s0 + tid];
    al[tid + 256] = attn[(size_t)b * S_ + s0 + tid + 256];
    __syncthreads();
    float a0 = 0.f, a1 = 0.f;
    const float* ep = enc + ((size_t)b * S_ + s0) * D_;
    for (int s = 0; s < 512; ++s) {
        const float a = al[s];
        a0 = fmaf(a, ep[(size_t)s * D_ + tid],       a0);
        a1 = fmaf(a, ep[(size_t)s * D_ + tid + 256], a1);
    }
    float* pp = part + (size_t)(b * 8 + c) * D_;
    pp[tid] = a0; pp[tid + 256] = a1;
}

// ---------------- kernel 5: deterministic partial reduce ----------------
__global__ __launch_bounds__(256) void ctx_reduce_kernel(
    const float* __restrict__ part, float* __restrict__ ctx)
{
    const int b = blockIdx.x, tid = threadIdx.x;
    float a0 = 0.f, a1 = 0.f;
    #pragma unroll
    for (int c = 0; c < 8; ++c) {
        const float* pp = part + (size_t)(b * 8 + c) * D_;
        a0 += pp[tid];
        a1 += pp[tid + 256];
    }
    ctx[(size_t)b * D_ + tid]       = a0;
    ctx[(size_t)b * D_ + tid + 256] = a1;
}

extern "C" void kernel_launch(void* const* d_in, const int* in_sizes, int n_in,
                              void* d_out, int out_size, void* d_ws, size_t ws_size,
                              hipStream_t stream) {
    const float* enc  = (const float*)d_in[0];
    const float* hdec = (const float*)d_in[1];
    const float* cdec = (const float*)d_in[2];
    const float* mask = (const float*)d_in[3];
    const float* cov  = (const float*)d_in[4];
    const float* Wh   = (const float*)d_in[5];
    const float* Ws   = (const float*)d_in[6];
    const float* Wsb  = (const float*)d_in[7];
    const float* wc   = (const float*)d_in[8];
    const float* vw   = (const float*)d_in[9];

    float* out  = (float*)d_out;
    float* ctx  = out;                 // (B, D)
    float* attn = out + B_ * D_;       // (B, S)
    float* covn = attn + B_ * S_;      // (B, S)

    float* ws    = (float*)d_ws;
    float* score = ws;                      // B*S floats
    float* dfeat = score + B_ * S_;         // B*D floats
    float* part  = dfeat + B_ * D_;         // B*8*D floats
    __bf16* whbf = (__bf16*)(part + B_ * 8 * D_); // D*D bf16

    whcvt_kernel<<<(D_ * D_) / (256 * 4), 256, 0, stream>>>(Wh, whbf);
    decfeat_kernel<<<B_, 256, 0, stream>>>(hdec, cdec, Ws, Wsb, dfeat);
    score_kernel<<<(B_ * S_) / 128, 256, 0, stream>>>(enc, whbf, dfeat, cov, wc, vw, score);
    softmax_kernel<<<B_, 256, 0, stream>>>(score, mask, cov, attn, covn);
    ctx_partial_kernel<<<dim3(B_, 8), 256, 0, stream>>>(attn, enc, part);
    ctx_reduce_kernel<<<B_, 256, 0, stream>>>(part, ctx);
}
